// SkimMemory_48713519071308
// MI455X (gfx1250) — compile-verified, matrix operand order checked by reading, not run on hardware
//
#include <hip/hip_runtime.h>
#include <hip/hip_bf16.h>
#include <stdint.h>

#define DIM   1024
#define CAP   32768
#define NQ    8192      // B*T = 4*2048
#define TOPK  32
#define NWAVES 4        // waves per block in the score kernel
#define TPW    2        // memory tiles per wave per sweep iteration

typedef __attribute__((ext_vector_type(16))) __bf16 v16bf;
typedef __attribute__((ext_vector_type(8)))  __bf16 v8bf;
typedef __attribute__((ext_vector_type(8)))  float  v8f;

union V16U { v16bf v; v8bf h[2]; };

// ---------------------------------------------------------------------------
// f32 -> bf16 conversion
// ---------------------------------------------------------------------------
__global__ __launch_bounds__(256) void f32_to_bf16_kernel(
    const float* __restrict__ in, __bf16* __restrict__ out, int n)
{
    int i = blockIdx.x * 256 + threadIdx.x;
    if (i < n) out[i] = (__bf16)in[i];
}

// ---------------------------------------------------------------------------
// Y[n][d] = sum_c X[n][c] * W[d][c]  (X: [rows x 1024] bf16, W: [1024x1024] bf16)
// Block: 256 threads = 8 waves, NO LDS, NO barriers.
// Per-wave tile: M=16 x N=64 (4 accumulators, A fragment reused 4x).
// Block tile: M=16 x N=512; grid = (rows/16, DIM/512).
// OUTF32 selects straight-line f32 vs bf16 stores (no runtime branch).
// ---------------------------------------------------------------------------
template <bool OUTF32>
__global__ __launch_bounds__(256) void gemm_xw_bf16(
    const __bf16* __restrict__ X, const __bf16* __restrict__ W,
    void* __restrict__ Yout)
{
    const int tid  = threadIdx.x;
    const int lane = tid & 31;
    const int wv   = tid >> 5;                 // 0..7
    const int ln   = lane & 15;
    const int half = (lane < 16) ? 0 : 8;      // ISA 16-bit fragment K-group offset
    const int row0 = blockIdx.x * 16;
    const int n0   = blockIdx.y * 512 + wv * 64;

    v8f acc[4] = {};
    for (int kc = 0; kc < DIM; kc += 32) {
        // A fragment straight from global (rows are L0/L2 resident across waves)
        V16U a;
        const __bf16* xa = X + (size_t)(row0 + ln) * DIM + kc + half;
        a.h[0] = *(const v8bf*)(xa);
        a.h[1] = *(const v8bf*)(xa + 16);
        #pragma unroll
        for (int j = 0; j < 4; ++j) {
            V16U b;
            const __bf16* wb = W + (size_t)(n0 + j * 16 + ln) * DIM + kc + half;
            b.h[0] = *(const v8bf*)(wb);
            b.h[1] = *(const v8bf*)(wb + 16);
            acc[j] = __builtin_amdgcn_wmma_f32_16x16x32_bf16(
                false, a.v, false, b.v, (short)0, acc[j], false, false);
        }
    }

    // D: VGPR r holds M = r + half, N = ln  (per 16-col sub-tile j)
    #pragma unroll
    for (int j = 0; j < 4; ++j) {
        #pragma unroll
        for (int r = 0; r < 8; ++r) {
            size_t idx = (size_t)(row0 + r + half) * DIM + (n0 + j * 16 + ln);
            if constexpr (OUTF32) ((float*)Yout)[idx]  = acc[j][r];
            else                  ((__bf16*)Yout)[idx] = (__bf16)acc[j][r];
        }
    }
}

// ---------------------------------------------------------------------------
// Fused: scores = Q·Kᵀ (WMMA), streaming top-32 per query row, softmax,
// context = attn · V[gather]  -> Ctx (bf16)
// Block: 128 threads = 4 waves, handles 16 query rows; each wave sweeps
// the 32768 memory rows, 2 16-wide tiles at a time (A fragment reused 2x).
// Kbf (64MB) is L2-resident (192MB L2) across the 512-block sweep.
// ---------------------------------------------------------------------------
__global__ __launch_bounds__(128) void scores_topk_ctx(
    const __bf16* __restrict__ Qbf, const __bf16* __restrict__ Kbf,
    const __bf16* __restrict__ Vbf, __bf16* __restrict__ Ctx)
{
    __shared__ __bf16 qt[16][DIM];                  // 32 KB: Q tile, block-resident
    __shared__ float  topS[NWAVES][16][TOPK];       // per-wave sorted top-32 (desc)
    __shared__ int    topI[NWAVES][16][TOPK];
    __shared__ float  stg[NWAVES][TPW][16][16];     // score-tile staging (8 KB)
    __shared__ float  finS[16][TOPK];
    __shared__ int    finI[16][TOPK];
    __shared__ float  attn[16][TOPK];

    const int tid  = threadIdx.x;
    const int lane = tid & 31;
    const int wv   = tid >> 5;                      // 0..3
    const int ln   = lane & 15;
    const int half = (lane < 16) ? 0 : 8;
    const int bt0  = blockIdx.x * 16;

    // load Q tile (16 x 1024 bf16) with 16B vector loads
    for (int i = tid; i < 16 * (DIM / 8); i += 128) {
        int m  = i >> 7;            // DIM/8 == 128
        int c8 = (i & 127) * 8;
        *(v8bf*)&qt[m][c8] = *(const v8bf*)(Qbf + (size_t)(bt0 + m) * DIM + c8);
    }
    // init top lists
    for (int i = tid; i < NWAVES * 16 * TOPK; i += 128) {
        ((float*)topS)[i] = -3.0e38f;
        ((int*)topI)[i]   = 0;
    }
    __syncthreads();

    // sweep memory rows: each wave takes 2 adjacent 16-row tiles per pass
    for (int t0 = wv * TPW; t0 < CAP / 16; t0 += NWAVES * TPW) {
        const int mt0 = t0 * 16;
        const int mt1 = mt0 + 16;
        v8f acc0 = {}, acc1 = {};
        for (int kc = 0; kc < DIM; kc += 32) {
            V16U a, b0, b1;
            a.h[0] = *(const v8bf*)&qt[ln][kc + half];
            a.h[1] = *(const v8bf*)&qt[ln][kc + half + 16];
            const __bf16* kb0 = Kbf + (size_t)(mt0 + ln) * DIM + kc + half;
            const __bf16* kb1 = Kbf + (size_t)(mt1 + ln) * DIM + kc + half;
            b0.h[0] = *(const v8bf*)(kb0);
            b0.h[1] = *(const v8bf*)(kb0 + 16);
            b1.h[0] = *(const v8bf*)(kb1);
            b1.h[1] = *(const v8bf*)(kb1 + 16);
            acc0 = __builtin_amdgcn_wmma_f32_16x16x32_bf16(
                false, a.v, false, b0.v, (short)0, acc0, false, false);
            acc1 = __builtin_amdgcn_wmma_f32_16x16x32_bf16(
                false, a.v, false, b1.v, (short)0, acc1, false, false);
        }
        // stage both 16x16 score tiles: D VGPR r -> row r+half, col ln
        #pragma unroll
        for (int r = 0; r < 8; ++r) {
            stg[wv][0][r + half][ln] = acc0[r];
            stg[wv][1][r + half][ln] = acc1[r];
        }
        __builtin_amdgcn_wave_barrier();   // HW keeps same-wave DS ops in order

        // lanes 0..15: sorted insertion of 32 candidates into this wave's top-32
        if (lane < 16) {
            const int rr = lane;
            #pragma unroll 1
            for (int n = 0; n < 32; ++n) {
                const int tt  = n >> 4;
                const int cc  = n & 15;
                float s = stg[wv][tt][rr][cc];
                if (s > topS[wv][rr][TOPK - 1]) {
                    int j = TOPK - 1;
                    while (j > 0 && topS[wv][rr][j - 1] < s) {
                        topS[wv][rr][j] = topS[wv][rr][j - 1];
                        topI[wv][rr][j] = topI[wv][rr][j - 1];
                        --j;
                    }
                    topS[wv][rr][j] = s;
                    topI[wv][rr][j] = mt0 + n;   // n covers 32 contiguous rows
                }
            }
        }
        __builtin_amdgcn_wave_barrier();
    }
    __syncthreads();

    // 4-way merge of sorted lists + softmax (1/sqrt(dim) folded in here;
    // scaling is monotonic so top-k order is unchanged)
    if (tid < 16) {
        const int rr = tid;
        int cur[NWAVES] = {0, 0, 0, 0};
        for (int o = 0; o < TOPK; ++o) {
            int bw = 0; float bs = -3.4e38f;
            #pragma unroll
            for (int w2 = 0; w2 < NWAVES; ++w2) {
                float s = (cur[w2] < TOPK) ? topS[w2][rr][cur[w2]] : -3.4e38f;
                if (s > bs) { bs = s; bw = w2; }
            }
            finS[rr][o] = bs;
            finI[rr][o] = topI[bw][rr][cur[bw]];
            cur[bw]++;
        }
        const float scale = 0.03125f;      // 1/sqrt(1024)
        float m0 = finS[rr][0];
        float sum = 0.f;
        #pragma unroll 1
        for (int o = 0; o < TOPK; ++o) sum += __expf((finS[rr][o] - m0) * scale);
        float inv = 1.0f / sum;
        #pragma unroll 1
        for (int o = 0; o < TOPK; ++o)
            attn[rr][o] = __expf((finS[rr][o] - m0) * scale) * inv;
    }
    __syncthreads();

    // context: Ctx[row][d0..d0+7] = sum_k attn[row][k] * V[idx[row][k]][d0..d0+7]
    // each of the 128 threads owns one 8-wide d-chunk (128*8 = 1024) -> b128 gathers
    {
        const int d0 = tid * 8;
        for (int rr = 0; rr < 16; ++rr) {
            float acc2[8] = {};
            #pragma unroll 1
            for (int o = 0; o < TOPK; ++o) {
                const float w = attn[rr][o];
                v8bf vv = *(const v8bf*)(Vbf + (size_t)finI[rr][o] * DIM + d0);
                #pragma unroll
                for (int e = 0; e < 8; ++e) acc2[e] += w * (float)vv[e];
            }
            v8bf ov;
            #pragma unroll
            for (int e = 0; e < 8; ++e) ov[e] = (__bf16)acc2[e];
            *(v8bf*)(Ctx + (size_t)(bt0 + rr) * DIM + d0) = ov;
        }
    }
}

// ---------------------------------------------------------------------------
// Host-side orchestration
// ---------------------------------------------------------------------------
extern "C" void kernel_launch(void* const* d_in, const int* in_sizes, int n_in,
                              void* d_out, int out_size, void* d_ws, size_t ws_size,
                              hipStream_t stream) {
    const float* query = (const float*)d_in[0];   // [4,2048,1024]
    const float* mem   = (const float*)d_in[1];   // [32768,1024]
    const float* Wq    = (const float*)d_in[2];
    const float* Wk    = (const float*)d_in[3];
    const float* Wv    = (const float*)d_in[4];
    const float* Wo    = (const float*)d_in[5];
    float* out = (float*)d_out;

    // workspace carve-up (256B aligned)
    size_t off = 0;
    auto carve = [&](size_t bytes) -> void* {
        void* p = (char*)d_ws + off;
        off += (bytes + 255) & ~(size_t)255;
        return p;
    };
    __bf16* q_bf   = (__bf16*)carve((size_t)NQ  * DIM * 2);
    __bf16* m_bf   = (__bf16*)carve((size_t)CAP * DIM * 2);
    __bf16* wq_bf  = (__bf16*)carve((size_t)DIM * DIM * 2);
    __bf16* wk_bf  = (__bf16*)carve((size_t)DIM * DIM * 2);
    __bf16* wv_bf  = (__bf16*)carve((size_t)DIM * DIM * 2);
    __bf16* wo_bf  = (__bf16*)carve((size_t)DIM * DIM * 2);
    __bf16* Qbf    = (__bf16*)carve((size_t)NQ  * DIM * 2);
    __bf16* Kbf    = (__bf16*)carve((size_t)CAP * DIM * 2);
    __bf16* Vbf    = (__bf16*)carve((size_t)CAP * DIM * 2);
    __bf16* Ctx    = (__bf16*)carve((size_t)NQ  * DIM * 2);

    auto cvt = [&](const float* src, __bf16* dst, int n) {
        f32_to_bf16_kernel<<<(n + 255) / 256, 256, 0, stream>>>(src, dst, n);
    };
    cvt(query, q_bf, NQ * DIM);
    cvt(mem,   m_bf, CAP * DIM);
    cvt(Wq, wq_bf, DIM * DIM);
    cvt(Wk, wk_bf, DIM * DIM);
    cvt(Wv, wv_bf, DIM * DIM);
    cvt(Wo, wo_bf, DIM * DIM);

    // projections: Q = query·Wqᵀ, K = mem·Wkᵀ, V = mem·Wvᵀ (bf16 out)
    gemm_xw_bf16<false><<<dim3(NQ / 16,  2), 256, 0, stream>>>(q_bf, wq_bf, Qbf);
    gemm_xw_bf16<false><<<dim3(CAP / 16, 2), 256, 0, stream>>>(m_bf, wk_bf, Kbf);
    gemm_xw_bf16<false><<<dim3(CAP / 16, 2), 256, 0, stream>>>(m_bf, wv_bf, Vbf);

    // fused scores + top-k + softmax + context
    scores_topk_ctx<<<dim3(NQ / 16), 128, 0, stream>>>(Qbf, Kbf, Vbf, Ctx);

    // out = Ctx·Woᵀ (f32 out)
    gemm_xw_bf16<true><<<dim3(NQ / 16, 2), 256, 0, stream>>>(Ctx, wo_bf, out);
}
